// Model_60902636257758
// MI455X (gfx1250) — compile-verified, matrix operand order checked by reading, not run on hardware
//
#include <hip/hip_runtime.h>
#include <hip/hip_bf16.h>
#include <math.h>

// ---------------- problem constants (match reference) ----------------
constexpr int NE  = 100000;   // email nodes
constexpr int NN  = 50000;    // noun nodes
constexpr int FE  = 256;      // email raw feature dim
constexpr int FN  = 128;      // noun raw feature dim
constexpr int HID = 128;      // hidden
constexpr int NH  = 4;        // heads
constexpr int HD  = 32;       // head dim
constexpr int EDG = 250000;   // edges per direction
constexpr int EL  = 100000;   // label edges
constexpr int NLAYER = 2;

typedef __attribute__((ext_vector_type(16))) __bf16 v16bf;
typedef __attribute__((ext_vector_type(8)))  float  v8f;

__device__ __forceinline__ float gelu_exact(float x) {
  return 0.5f * x * (1.0f + erff(x * 0.7071067811865475f));
}

// ---------------- tiny utility kernels ----------------
__global__ void k_fill(float* __restrict__ p, float v, size_t n) {
  size_t i = (size_t)blockIdx.x * blockDim.x + threadIdx.x;
  if (i < n) p[i] = v;
}

// Expand per-head [4,32,32] relation matrix into block-diagonal [128,128]
__global__ void k_blockdiag(const float* __restrict__ rel, float* __restrict__ bd) {
  int i = blockIdx.x * blockDim.x + threadIdx.x;   // 0..16383
  if (i >= HID * HID) return;
  int r = i >> 7, c = i & 127;
  int hr = r >> 5, hc = c >> 5;
  bd[i] = (hr == hc) ? rel[hr * (HD * HD) + (r & 31) * HD + (c & 31)] : 0.0f;
}

// ---------------- WMMA GEMM: C[M,128] = act(A[M,K] @ W[K,128] + b) ----------------
// Weights pre-packed into LDS in WMMA B-fragment order: row (ct,ks,lane) holds the
// 16 bf16 elements lane needs -> fragment load = one 32B v16bf read (2x ds_load_b128).
// Fragment element i maps to K-offset (i<8 ? i : i+8) relative to (ks*32 + khalf):
// this makes both A and B staging a pair of contiguous 8-element chunks.
template <int K, int ACT_IN>
__launch_bounds__(256)
__global__ void k_gemm_wmma(const float* __restrict__ A, const float* __restrict__ W,
                            const float* __restrict__ bias, float* __restrict__ C,
                            int M, int act_out) {
  constexpr int KS = K / 32;                        // 32-deep k-steps
  __shared__ v16bf sB[8 * KS * 32];                 // 8 col-tiles x KS x 32 lanes (32B each)

  const int tid = threadIdx.x;

  // ---- stage weights, packed per-fragment ----
  for (int r = tid; r < 8 * KS * 32; r += 256) {
    int ct  = r / (KS * 32);
    int rem = r - ct * (KS * 32);
    int ks  = rem >> 5;
    int L   = rem & 31;
    int col   = ct * 16 + (L & 15);
    int kbase = ks * 32 + ((L >> 4) << 3);          // +8 for lanes 16..31
    v16bf t;
#pragma unroll
    for (int i = 0; i < 16; ++i) {
      int kk = kbase + (i < 8 ? i : i + 8);
      t[i] = (__bf16)W[kk * HID + col];
    }
    sB[r] = t;
  }
  __syncthreads();

  const int wave = tid >> 5;
  const int lane = tid & 31;
  const int row0 = blockIdx.x * 128 + wave * 16;
  if (row0 >= M) return;

  v8f acc[8] = {};

  const int  mrow  = row0 + (lane & 15);
  const int  khalf = (lane >> 4) << 3;              // 0 or 8
  const bool rowok = (mrow < M);
  const float* Arow = A + (size_t)mrow * K;

#pragma unroll
  for (int ks = 0; ks < KS; ++ks) {
    const int k0 = ks * 32;
    // A fragment: two contiguous 32B chunks -> 4x global_load_b128
    float4 c0, c1, c2, c3;
    if (rowok) {
      const float4* p0 = (const float4*)(Arow + k0 + khalf);
      const float4* p1 = (const float4*)(Arow + k0 + 16 + khalf);
      c0 = p0[0]; c1 = p0[1]; c2 = p1[0]; c3 = p1[1];
    } else {
      c0 = c1 = c2 = c3 = make_float4(0.f, 0.f, 0.f, 0.f);
    }
    float av[16] = { c0.x, c0.y, c0.z, c0.w, c1.x, c1.y, c1.z, c1.w,
                     c2.x, c2.y, c2.z, c2.w, c3.x, c3.y, c3.z, c3.w };
    v16bf afrag;
#pragma unroll
    for (int i = 0; i < 16; ++i) {
      float x = av[i];
      if (ACT_IN) x = gelu_exact(x);
      afrag[i] = (__bf16)x;
    }
#pragma unroll
    for (int ct = 0; ct < 8; ++ct) {
      v16bf bfrag = sB[(ct * KS + ks) * 32 + lane];
      acc[ct] = __builtin_amdgcn_wmma_f32_16x16x32_bf16(
          false, afrag, false, bfrag, (short)0, acc[ct], false, false);
    }
  }

  // Epilogue: C/D layout — VGPR v holds row (v + 8*(lane>=16)), col = lane&15
  const int rbase = row0 + ((lane >> 4) << 3);
  const int ncol  = lane & 15;
#pragma unroll
  for (int ct = 0; ct < 8; ++ct) {
    int col = ct * 16 + ncol;
    float bv = bias ? bias[col] : 0.0f;
#pragma unroll
    for (int v = 0; v < 8; ++v) {
      int row = rbase + v;
      if (row < M) {
        float x = acc[ct][v] + bv;
        if (act_out == 1) x = fmaxf(x, 0.0f);
        C[(size_t)row * HID + col] = x;
      }
    }
  }
}

// ---------------- edge-phase kernels ----------------
__device__ __forceinline__ void atomicMaxF(float* addr, float val) {
  if (val >= 0.0f) atomicMax((int*)addr, __float_as_int(val));
  else             atomicMin((unsigned int*)addr, (unsigned int)__float_as_int(val));
}

// pass 1: alpha[e,h] = <q[dst,h,:], kr[src,h,:]> * prel[h] / sqrt(D); seg-max into mbuf
__global__ void k_edge_alpha(const int* __restrict__ src, const int* __restrict__ dst,
                             const float* __restrict__ q, const float* __restrict__ kr,
                             const float* __restrict__ prel,
                             float* __restrict__ alpha, float* __restrict__ mbuf) {
  int gid = blockIdx.x * blockDim.x + threadIdx.x;
  if (gid >= EDG * NH) return;
  int e = gid >> 2, h = gid & 3;
  int s = src[e], d = dst[e];
  const float4* qp = (const float4*)(q  + (size_t)d * HID + h * HD);
  const float4* kp = (const float4*)(kr + (size_t)s * HID + h * HD);
  float acc = 0.0f;
#pragma unroll
  for (int i = 0; i < HD / 4; ++i) {
    float4 a = qp[i], b = kp[i];
    acc = fmaf(a.x, b.x, acc); acc = fmaf(a.y, b.y, acc);
    acc = fmaf(a.z, b.z, acc); acc = fmaf(a.w, b.w, acc);
  }
  float al = acc * prel[h] * 0.1767766952966369f;   // 1/sqrt(32)
  alpha[gid] = al;
  atomicMaxF(&mbuf[d * NH + h], al);
}

// pass 2: alpha <- exp(alpha - m[dst]); seg-sum into sbuf
__global__ void k_edge_expsum(const int* __restrict__ dst, float* __restrict__ alpha,
                              const float* __restrict__ mbuf, float* __restrict__ sbuf) {
  int gid = blockIdx.x * blockDim.x + threadIdx.x;
  if (gid >= EDG * NH) return;
  int e = gid >> 2, h = gid & 3;
  int d = dst[e];
  float m = mbuf[d * NH + h];
  if (!(m > -3.0e38f)) m = 0.0f;                    // empty-segment guard (matches ref)
  float a = __expf(alpha[gid] - m);
  alpha[gid] = a;
  atomicAdd(&sbuf[d * NH + h], a);
}

// pass 3: agg[dst] += vr[src] * alpha/(s[dst]+eps); one thread per (edge, 4 channels)
__global__ void k_edge_msg(const int* __restrict__ src, const int* __restrict__ dst,
                           const float* __restrict__ alpha, const float* __restrict__ sbuf,
                           const float* __restrict__ vr, float* __restrict__ agg) {
  size_t gid = (size_t)blockIdx.x * blockDim.x + threadIdx.x;
  if (gid >= (size_t)EDG * (HID / 4)) return;
  int e  = (int)(gid >> 5);
  int c4 = (int)(gid & 31);                         // float4 index within row
  int c  = c4 * 4;
  int h  = c >> 5;
  int s = src[e], d = dst[e];
  float a = alpha[e * NH + h] / (sbuf[d * NH + h] + 1e-16f);
  float4 v = ((const float4*)(vr + (size_t)s * HID))[c4];
  float* ap = agg + (size_t)d * HID + c;
  atomicAdd(ap + 0, v.x * a);
  atomicAdd(ap + 1, v.y * a);
  atomicAdd(ap + 2, v.z * a);
  atomicAdd(ap + 3, v.w * a);
}

// skip mixing: x = sigmoid(skip)*o + (1-sigmoid(skip))*x   (float4, in place)
__global__ void k_skipmix(const float* __restrict__ o, float* __restrict__ x,
                          const float* __restrict__ skip, size_t n4) {
  size_t i = (size_t)blockIdx.x * blockDim.x + threadIdx.x;
  if (i >= n4) return;
  float s = 1.0f / (1.0f + __expf(-skip[0]));
  float t = 1.0f - s;
  float4 ov = ((const float4*)o)[i];
  float4 xv = ((float4*)x)[i];
  xv.x = s * ov.x + t * xv.x; xv.y = s * ov.y + t * xv.y;
  xv.z = s * ov.z + t * xv.z; xv.w = s * ov.w + t * xv.w;
  ((float4*)x)[i] = xv;
}

// classifier: pred[i] = <xe[ea[i]], xn[eb[i]]>
__global__ void k_classify(const float* __restrict__ xe, const float* __restrict__ xn,
                           const int* __restrict__ ea, const int* __restrict__ eb,
                           float* __restrict__ out) {
  int i = blockIdx.x * blockDim.x + threadIdx.x;
  if (i >= EL) return;
  const float4* pa = (const float4*)(xe + (size_t)ea[i] * HID);
  const float4* pb = (const float4*)(xn + (size_t)eb[i] * HID);
  float acc = 0.0f;
#pragma unroll
  for (int c = 0; c < HID / 4; ++c) {
    float4 a = pa[c], b = pb[c];
    acc = fmaf(a.x, b.x, acc); acc = fmaf(a.y, b.y, acc);
    acc = fmaf(a.z, b.z, acc); acc = fmaf(a.w, b.w, acc);
  }
  out[i] = acc;
}

// ---------------- host orchestration ----------------
extern "C" void kernel_launch(void* const* d_in, const int* in_sizes, int n_in,
                              void* d_out, int out_size, void* d_ws, size_t ws_size,
                              hipStream_t stream) {
  (void)in_sizes; (void)n_in; (void)out_size; (void)ws_size;

  const float* x_email = (const float*)d_in[0];
  const float* x_noun  = (const float*)d_in[1];
  const int* e2n_src = (const int*)d_in[2];
  const int* e2n_dst = e2n_src + EDG;
  const int* n2e_src = (const int*)d_in[3];
  const int* n2e_dst = n2e_src + EDG;
  const int* el_e = (const int*)d_in[4];
  const int* el_n = el_e + EL;
  auto P = [&](int i) { return (const float*)d_in[i]; };

  // workspace layout (floats)
  float* w = (float*)d_ws;
  size_t off = 0;
  auto alloc = [&](size_t n) { float* p = w + off; off += n; return p; };
  float* xe   = alloc((size_t)NE * HID);
  float* xn   = alloc((size_t)NN * HID);
  float* ke   = alloc((size_t)NE * HID);
  float* qe   = alloc((size_t)NE * HID);
  float* ve   = alloc((size_t)NE * HID);
  float* kn   = alloc((size_t)NN * HID);
  float* qn   = alloc((size_t)NN * HID);
  float* vn   = alloc((size_t)NN * HID);
  float* kre  = alloc((size_t)NE * HID);
  float* vre  = alloc((size_t)NE * HID);
  float* krn  = alloc((size_t)NN * HID);
  float* vrn  = alloc((size_t)NN * HID);
  float* agge = alloc((size_t)NE * HID);
  float* aggn = alloc((size_t)NN * HID);
  float* alpha= alloc((size_t)EDG * NH);
  float* mbuf = alloc((size_t)NE * NH);
  float* sbuf = alloc((size_t)NE * NH);
  float* bd   = alloc((size_t)HID * HID);

  auto gemm = [&](const float* A, const float* W_, const float* b, float* C,
                  int M, int K, int ai, int ao) {
    dim3 grid((M + 127) / 128), blk(256);
    if (K == 256) {
      k_gemm_wmma<256, 0><<<grid, blk, 0, stream>>>(A, W_, b, C, M, ao);
    } else if (ai == 1) {
      k_gemm_wmma<128, 1><<<grid, blk, 0, stream>>>(A, W_, b, C, M, ao);
    } else {
      k_gemm_wmma<128, 0><<<grid, blk, 0, stream>>>(A, W_, b, C, M, ao);
    }
  };
  auto fill = [&](float* p, float v, size_t n) {
    k_fill<<<dim3((unsigned)((n + 255) / 256)), 256, 0, stream>>>(p, v, n);
  };
  auto blockdiag = [&](const float* rel) {
    k_blockdiag<<<dim3(64), dim3(256), 0, stream>>>(rel, bd);
  };

  // -------- input projections (ReLU) --------
  gemm(x_email, P(5), P(6), xe, NE, FE, 0, 1);
  gemm(x_noun,  P(7), P(8), xn, NN, FN, 0, 1);

  const int EH_BLK  = (EDG * NH + 255) / 256;
  const int MSG_BLK = (int)(((size_t)EDG * (HID / 4) + 255) / 256);

  for (int l = 0; l < NLAYER; ++l) {
    const int b0 = 9 + l * 24;

    // K/Q/V projections, both node types
    gemm(xe, P(b0 + 0),  P(b0 + 1),  ke, NE, HID, 0, 0);
    gemm(xe, P(b0 + 2),  P(b0 + 3),  qe, NE, HID, 0, 0);
    gemm(xe, P(b0 + 4),  P(b0 + 5),  ve, NE, HID, 0, 0);
    gemm(xn, P(b0 + 9),  P(b0 + 10), kn, NN, HID, 0, 0);
    gemm(xn, P(b0 + 11), P(b0 + 12), qn, NN, HID, 0, 0);
    gemm(xn, P(b0 + 13), P(b0 + 14), vn, NN, HID, 0, 0);

    fill(agge, 0.0f, (size_t)NE * HID);
    fill(aggn, 0.0f, (size_t)NN * HID);

    // ---- relation e2n: src = email, dst = noun ----
    blockdiag(P(b0 + 18));                       // arel_e2n
    gemm(ke, bd, nullptr, kre, NE, HID, 0, 0);
    blockdiag(P(b0 + 19));                       // mrel_e2n
    gemm(ve, bd, nullptr, vre, NE, HID, 0, 0);
    fill(mbuf, -INFINITY, (size_t)NN * NH);
    fill(sbuf, 0.0f,      (size_t)NN * NH);
    k_edge_alpha <<<EH_BLK, 256, 0, stream>>>(e2n_src, e2n_dst, qn, kre, P(b0 + 20), alpha, mbuf);
    k_edge_expsum<<<EH_BLK, 256, 0, stream>>>(e2n_dst, alpha, mbuf, sbuf);
    k_edge_msg   <<<MSG_BLK, 256, 0, stream>>>(e2n_src, e2n_dst, alpha, sbuf, vre, aggn);

    // ---- relation n2e: src = noun, dst = email ----
    blockdiag(P(b0 + 21));                       // arel_n2e
    gemm(kn, bd, nullptr, krn, NN, HID, 0, 0);
    blockdiag(P(b0 + 22));                       // mrel_n2e
    gemm(vn, bd, nullptr, vrn, NN, HID, 0, 0);
    fill(mbuf, -INFINITY, (size_t)NE * NH);
    fill(sbuf, 0.0f,      (size_t)NE * NH);
    k_edge_alpha <<<EH_BLK, 256, 0, stream>>>(n2e_src, n2e_dst, qe, krn, P(b0 + 23), alpha, mbuf);
    k_edge_expsum<<<EH_BLK, 256, 0, stream>>>(n2e_dst, alpha, mbuf, sbuf);
    k_edge_msg   <<<MSG_BLK, 256, 0, stream>>>(n2e_src, n2e_dst, alpha, sbuf, vrn, agge);

    // output projection with GELU on input; reuse ke/kn as 'o'
    gemm(agge, P(b0 + 6),  P(b0 + 7),  ke, NE, HID, 1, 0);
    gemm(aggn, P(b0 + 15), P(b0 + 16), kn, NN, HID, 1, 0);

    size_t ne4 = (size_t)NE * HID / 4, nn4 = (size_t)NN * HID / 4;
    k_skipmix<<<dim3((unsigned)((ne4 + 255) / 256)), 256, 0, stream>>>(ke, xe, P(b0 + 8),  ne4);
    k_skipmix<<<dim3((unsigned)((nn4 + 255) / 256)), 256, 0, stream>>>(kn, xn, P(b0 + 17), nn4);
  }

  // -------- classifier --------
  k_classify<<<dim3((EL + 255) / 256), 256, 0, stream>>>(xe, xn, el_e, el_n, (float*)d_out);
}
